// SoftInstanceLossBoost_88072599372560
// MI455X (gfx1250) — compile-verified, matrix operand order checked by reading, not run on hardware
//
#include <hip/hip_runtime.h>
#include <hip/hip_bf16.h>
#include <hip/hip_fp16.h>

// ---------------------------------------------------------------------------
// SoftInstanceLoss (boosting/pull-only) for MI455X (gfx1250, wave32, WMMA).
//
// Pipeline:
//   K0: zero the scalar output
//   K1: sq[r] = ||F[r]||^2                       (F = concat(z_i, z_j), [4096,512])
//   K2: pd = sqrt(max(sq_i + sq_j - 2*F F^T, 0)) via bf16x3-split WMMA GEMM
//   K3: per row: 10 smallest off-diag distances (== top-10 of w, since w is
//       monotone in pd), radius = smallest; loss += w * mask * <probs_i,probs_j>
//
// Workspace layout: [0,16KB) = sq (4096 f32); [16KB, 16KB+64MB) = pd.
// ---------------------------------------------------------------------------

#ifndef BB
#define BB 4096   // 2n
#define NN 2048   // n
#define DD 512    // feature dim
#define CC 128    // classes
#endif

typedef float  v8f   __attribute__((ext_vector_type(8)));
typedef __bf16 v16bf __attribute__((ext_vector_type(16)));
typedef __bf16 bf16x8 __attribute__((ext_vector_type(8)));
typedef __bf16 bf16x4 __attribute__((ext_vector_type(4)));

__device__ __forceinline__ const float* Frow(const float* zi, const float* zj, int r) {
  return (r < NN) ? (zi + (size_t)r * DD) : (zj + (size_t)(r - NN) * DD);
}
__device__ __forceinline__ const float* Prow(const float* pi, const float* pj, int r) {
  return (r < NN) ? (pi + (size_t)r * CC) : (pj + (size_t)(r - NN) * CC);
}

// ---------------------------------------------------------------- K0: init out
__global__ void zero_out_kernel(float* out) { out[0] = 0.0f; }

// ---------------------------------------------------------------- K1: norms
__global__ __launch_bounds__(256) void row_norm_kernel(
    const float* __restrict__ zi, const float* __restrict__ zj,
    float* __restrict__ sq) {
  const int lane = threadIdx.x & 31;
  const int row  = blockIdx.x * 8 + (threadIdx.x >> 5);
  const float* p = Frow(zi, zj, row);
  float s = 0.0f;
#pragma unroll
  for (int t = 0; t < DD / 32; ++t) {
    float v = p[lane + 32 * t];
    s += v * v;
  }
#pragma unroll
  for (int off = 16; off; off >>= 1) s += __shfl_xor(s, off, 32);
  if (lane == 0) sq[row] = s;
}

// ---------------------------------------------------------------- K2: GEMM+pd
// LDS panels: 128 rows x 32 k, bf16 hi + lo, row stride 40 bf16 (80 bytes):
// 16B-aligned rows, and the 16B fragment reads hit 64 distinct banks.
constexpr int LDST = 40;

__device__ __forceinline__ void split_store(__bf16* hi, __bf16* lo, float4 f) {
  __bf16 h0 = (__bf16)f.x, h1 = (__bf16)f.y, h2 = (__bf16)f.z, h3 = (__bf16)f.w;
  bf16x4 hv = {h0, h1, h2, h3};
  bf16x4 lv = {(__bf16)(f.x - (float)h0), (__bf16)(f.y - (float)h1),
               (__bf16)(f.z - (float)h2), (__bf16)(f.w - (float)h3)};
  *(bf16x4*)hi = hv;
  *(bf16x4*)lo = lv;
}

// A/B 16-bit fragment per ISA layout: lane holds K[8h..8h+7] (elems 0-7) and
// K[16+8h..16+8h+7] (elems 8-15), h = lane/16, row = base + lane%16.
__device__ __forceinline__ v16bf load_frag(const __bf16* rowbase, int h) {
  union { v16bf v; bf16x8 q[2]; } u;
  u.q[0] = *(const bf16x8*)(rowbase + 8 * h);
  u.q[1] = *(const bf16x8*)(rowbase + 16 + 8 * h);
  return u.v;
}

__global__ __launch_bounds__(256) void gram_pd_kernel(
    const float* __restrict__ zi, const float* __restrict__ zj,
    const float* __restrict__ sq, float* __restrict__ pd) {
  __shared__ __bf16 sAhi[128 * LDST];
  __shared__ __bf16 sAlo[128 * LDST];
  __shared__ __bf16 sBhi[128 * LDST];
  __shared__ __bf16 sBlo[128 * LDST];

  const int tid  = threadIdx.x;
  const int lane = tid & 31;
  const int wid  = tid >> 5;       // 0..7
  const int l16  = lane & 15;
  const int h    = lane >> 4;      // half-wave
  const int wm   = wid >> 2;       // 0..1 : 64-row sub-panel
  const int wn   = wid & 3;        // 0..3 : 32-col sub-panel
  const int tileM = blockIdx.y * 128;
  const int tileN = blockIdx.x * 128;

  v8f acc[4][2];
#pragma unroll
  for (int mt = 0; mt < 4; ++mt)
#pragma unroll
    for (int nt = 0; nt < 2; ++nt) acc[mt][nt] = (v8f){};

  for (int k0 = 0; k0 < DD; k0 += 32) {
    __syncthreads();
    // Stage 128x32 f32 per side -> bf16 hi/lo panels. 1024 float4 per panel,
    // 4 per thread.
#pragma unroll
    for (int i = 0; i < 4; ++i) {
      const int fid = tid + (i << 8);
      const int row = fid >> 3;
      const int seg = fid & 7;
      const float4 fa = *(const float4*)(Frow(zi, zj, tileM + row) + k0 + seg * 4);
      split_store(&sAhi[row * LDST + seg * 4], &sAlo[row * LDST + seg * 4], fa);
      const float4 fb = *(const float4*)(Frow(zi, zj, tileN + row) + k0 + seg * 4);
      split_store(&sBhi[row * LDST + seg * 4], &sBlo[row * LDST + seg * 4], fb);
    }
    __syncthreads();

    v16bf ah[4], al[4], bh[2], bl[2];
#pragma unroll
    for (int mt = 0; mt < 4; ++mt) {
      const int r = wm * 64 + mt * 16 + l16;
      ah[mt] = load_frag(&sAhi[r * LDST], h);
      al[mt] = load_frag(&sAlo[r * LDST], h);
    }
#pragma unroll
    for (int nt = 0; nt < 2; ++nt) {
      const int r = wn * 32 + nt * 16 + l16;
      bh[nt] = load_frag(&sBhi[r * LDST], h);
      bl[nt] = load_frag(&sBlo[r * LDST], h);
    }

    // bf16x3 split GEMM: G ~= hi*hi^T + hi*lo^T + lo*hi^T (~f32 accuracy)
#pragma unroll
    for (int mt = 0; mt < 4; ++mt)
#pragma unroll
      for (int nt = 0; nt < 2; ++nt) {
        acc[mt][nt] = __builtin_amdgcn_wmma_f32_16x16x32_bf16(
            false, ah[mt], false, bh[nt], (short)0, acc[mt][nt], false, false);
        acc[mt][nt] = __builtin_amdgcn_wmma_f32_16x16x32_bf16(
            false, ah[mt], false, bl[nt], (short)0, acc[mt][nt], false, false);
        acc[mt][nt] = __builtin_amdgcn_wmma_f32_16x16x32_bf16(
            false, al[mt], false, bh[nt], (short)0, acc[mt][nt], false, false);
      }
  }

  // Epilogue: d2 = sq[M] + sq[N] - 2G, pd = sqrt(max(d2,0)).
  // C/D layout: VGPR e -> M = e + 8*(lane/16), N = lane%16.
#pragma unroll
  for (int mt = 0; mt < 4; ++mt) {
    const int Mb = tileM + wm * 64 + mt * 16 + 8 * h;
#pragma unroll
    for (int nt = 0; nt < 2; ++nt) {
      const int N = tileN + wn * 32 + nt * 16 + l16;
      const float sn = sq[N];
#pragma unroll
      for (int e = 0; e < 8; ++e) {
        const int M = Mb + e;
        const float d2 = sq[M] + sn - 2.0f * acc[mt][nt][e];
        pd[(size_t)M * BB + N] = __builtin_sqrtf(fmaxf(d2, 0.0f));
      }
    }
  }
}

// ---------------------------------------------------------------- K3: kNN+loss
__global__ __launch_bounds__(256) void topk_loss_kernel(
    const float* __restrict__ pd,
    const float* __restrict__ pi, const float* __restrict__ pj,
    const int* __restrict__ label, float* __restrict__ out) {
  const int lane = threadIdx.x & 31;
  const int row  = blockIdx.x * 8 + (threadIdx.x >> 5);
  const float FINF = __builtin_inff();

  // Preload this row's prob slice (4 values per lane, stride 32).
  const float* prow = Prow(pi, pj, row);
  float pr[4];
#pragma unroll
  for (int t = 0; t < 4; ++t) pr[t] = prow[lane + 32 * t];

  // Lane-local sorted top-10 smallest (value, index), tie-break smaller index
  // to match jax.lax.top_k's stable tie handling on the (monotone) weights.
  float val[10];
  int   idx[10];
#pragma unroll
  for (int k = 0; k < 10; ++k) { val[k] = FINF; idx[k] = 0x7fffffff; }

  const float* prd = pd + (size_t)row * BB;
  for (int j = lane; j < BB; j += 32) {
    if (j == row) continue;  // exclude diagonal (w there is forced to 0)
    float v = prd[j];
    if (v < val[9] || (v == val[9] && j < idx[9])) {
      float cv = v; int ci = j;
#pragma unroll
      for (int k = 0; k < 10; ++k) {
        bool ins = (cv < val[k]) || (cv == val[k] && ci < idx[k]);
        float tv = ins ? cv : val[k];
        int   ti = ins ? ci : idx[k];
        if (ins) { cv = val[k]; ci = idx[k]; }
        val[k] = tv; idx[k] = ti;
      }
    }
  }

  // Wave-wide merge: 10 rounds of butterfly argmin with index tie-break.
  float wv[10]; int wi[10];
#pragma unroll
  for (int s = 0; s < 10; ++s) {
    float bv = val[0]; int bi = idx[0];
#pragma unroll
    for (int off = 16; off; off >>= 1) {
      float ov = __shfl_xor(bv, off, 32);
      int   oi = __shfl_xor(bi, off, 32);
      if (ov < bv || (ov == bv && oi < bi)) { bv = ov; bi = oi; }
    }
    wv[s] = bv; wi[s] = bi;
    if (idx[0] == bi) {  // winner owner pops (indices are unique)
#pragma unroll
      for (int k = 0; k < 9; ++k) { val[k] = val[k + 1]; idx[k] = idx[k + 1]; }
      val[9] = FINF; idx[9] = 0x7fffffff;
    }
  }

  const float radius = wv[0];        // 1-NN distance (sorted[:,1] in reference)
  const float inv_r  = 1.0f / radius;
  const int a  = row & (NN - 1);
  const int la = label[a];

  float part = 0.0f;                  // lane-local partial of sum_k w_k * <p_i,p_j>
#pragma unroll
  for (int s = 0; s < 10; ++s) {
    const int j = wi[s];
    const int b = j & (NN - 1);
    const float w =
        1.0f - fminf(fmaxf((wv[s] - radius) * inv_r, 0.0f), 1.0f);
    const bool m = (a != b) && (la == label[b]) && (la != -1);
    if (m && w > 0.0f) {
      const float* pn = Prow(pi, pj, j);
      float dp = 0.0f;
#pragma unroll
      for (int t = 0; t < 4; ++t) dp += pr[t] * pn[lane + 32 * t];
      part += w * dp;
    }
  }
#pragma unroll
  for (int off = 16; off; off >>= 1) part += __shfl_xor(part, off, 32);
  if (lane == 0)
    atomicAdd(out, part * (1.0f / ((float)BB * (float)BB)));
}

// ---------------------------------------------------------------------------
extern "C" void kernel_launch(void* const* d_in, const int* in_sizes, int n_in,
                              void* d_out, int out_size, void* d_ws, size_t ws_size,
                              hipStream_t stream) {
  const float* zi  = (const float*)d_in[0];   // [2048,512]
  const float* zj  = (const float*)d_in[1];   // [2048,512]
  const float* pip = (const float*)d_in[2];   // [2048,128]
  const float* pjp = (const float*)d_in[3];   // [2048,128]
  const int*   lab = (const int*)d_in[4];     // [2048]
  float* out = (float*)d_out;

  // ws: [0,16KB) sq, [16KB, 16KB+64MB) pd
  float* sq = (float*)d_ws;
  float* pd = (float*)((char*)d_ws + 16384);

  zero_out_kernel<<<1, 1, 0, stream>>>(out);
  row_norm_kernel<<<BB / 8, 256, 0, stream>>>(zi, zj, sq);
  gram_pd_kernel<<<dim3(BB / 128, BB / 128), 256, 0, stream>>>(zi, zj, sq, pd);
  topk_loss_kernel<<<BB / 8, 256, 0, stream>>>(pd, pip, pjp, lab, out);
}
